// SelectiveSSMLayer_45655502356556
// MI455X (gfx1250) — matrix operand order repaired and checked by reading, not compiled, not run on hardware
//
#include <hip/hip_runtime.h>
#include <math.h>

// ---------------- problem constants ----------------
#define DM    1024      // d_model
#define DI    2048      // d_inner
#define NS    16        // n_state
#define DR    64        // dt_rank
#define BB    2         // batch
#define SS    2048      // seq
#define MROWS (BB*SS)   // 4096 rows (b*l flattened)
#define EPSL  1e-5f
#define KT    64        // k-chunk staged in LDS per TDM transfer

#if defined(__has_builtin)
#  if __has_builtin(__builtin_amdgcn_tensor_load_to_lds) && \
      __has_builtin(__builtin_amdgcn_s_wait_tensorcnt)
#    define USE_TDM 1
#  else
#    define USE_TDM 0
#  endif
#else
#  define USE_TDM 0
#endif

// ---------------- CDNA5 WMMA / TDM types ----------------
typedef __attribute__((ext_vector_type(16))) __bf16        v16bf;
typedef __attribute__((ext_vector_type(8)))  __bf16        v8bf;
typedef __attribute__((ext_vector_type(8)))  float         v8f;
typedef __attribute__((ext_vector_type(4)))  unsigned int  v4u;
typedef __attribute__((ext_vector_type(8)))  int           v8i_t;
typedef __attribute__((ext_vector_type(4)))  int           v4i_t;

__device__ __forceinline__ v8f wmma_bf16(v16bf a, v16bf b, v8f c) {
    // D = A(16x32 bf16) * B(32x16 bf16) + C(16x16 f32)
    return __builtin_amdgcn_wmma_f32_16x16x32_bf16(
        false, a, false, b, (short)0, c, false, false);
}

#if USE_TDM
// Tensor Data Mover: DMA a 2-D bf16 tile (tile_d1 rows x tile_d0 elems, row
// stride `stride0` elems) from global into LDS at byte offset `lds_addr`.
// D# layout per CDNA5 ISA (group0/group1; groups 2-3 and
// the trailing group zero for 2-D tiles: tile_dim2/3 = 0 -> unused).
// NOTE: this toolchain exposes the 6-arg builtin
//   (uint32x4 g0, int32x8 g1, int32x4, int32x4, int32x8, i32 cpol).
__device__ __forceinline__ void tdm_load_tile(unsigned lds_addr, const __bf16* gptr,
                                              unsigned tensor_d0, unsigned tensor_d1,
                                              unsigned tile_d0, unsigned tile_d1,
                                              unsigned stride0) {
    unsigned long long ga = (unsigned long long)(size_t)gptr;
    v4u g0;
    g0[0] = 1u;                                            // count=1, user mode
    g0[1] = lds_addr;                                      // lds_addr [63:32]
    g0[2] = (unsigned)ga;                                  // global_addr [95:64]
    g0[3] = (unsigned)((ga >> 32) & 0x01FFFFFFu)           // global_addr [120:96]
          | (2u << 30);                                    // type=2 ("image")
    v8i_t g1;
    g1[0] = (int)(1u << 16);                               // data_size=1 -> 2 bytes
    g1[1] = (int)((tensor_d0 & 0xFFFFu) << 16);            // tensor_dim0[15:0] @bit48
    g1[2] = (int)(((tensor_d0 >> 16) & 0xFFFFu)            // tensor_dim0[31:16]
          | ((tensor_d1 & 0xFFFFu) << 16));                // tensor_dim1[15:0]
    g1[3] = (int)(((tensor_d1 >> 16) & 0xFFFFu)            // tensor_dim1[31:16]
          | ((tile_d0 & 0xFFFFu) << 16));                  // tile_dim0
    g1[4] = (int)(tile_d1 & 0xFFFFu);                      // tile_dim1 (tile_dim2=0)
    g1[5] = (int)stride0;                                  // tensor_dim0_stride[31:0]
    g1[6] = 0;                                             // stride hi / dim1_stride
    g1[7] = 0;
    v4i_t z4 = {0, 0, 0, 0};
    v8i_t z8 = {0, 0, 0, 0, 0, 0, 0, 0};
    __builtin_amdgcn_tensor_load_to_lds(g0, g1, z4, z4, z8, 0);
}
#endif

// Cooperative fallback copy (used when TDM builtin is unavailable).
__device__ __forceinline__ void copy_tile_coop(__bf16* dst, const __bf16* src,
                                               int rows, int rowlen, int srcstride,
                                               int tid, int nthreads) {
    int vecs = rows * (rowlen >> 3);                       // 8 bf16 = 16B per vec
    int vpr  = rowlen >> 3;
    for (int i = tid; i < vecs; i += nthreads) {
        int r = i / vpr, c = (i - r * vpr) << 3;
        *(v8bf*)(dst + r * rowlen + c) = *(const v8bf*)(src + (size_t)r * srcstride + c);
    }
}

// A-fragment (16x32, row-major MxK), converting f32 -> bf16 on the fly.
// Per ISA 7.12.2: lane row = lane&15; khalf = lane>>4; dwords 0-3 hold K=khalf*8..+7,
// dwords 4-7 hold K=16+khalf*8..+7 (relative to the 32-wide k-tile).
__device__ __forceinline__ v16bf load_a_f32frag(const float* __restrict__ A, int ldk,
                                                int row, int kt, int lane) {
    const float* p = A + (size_t)row * ldk + kt + ((lane >> 4) << 3);
    float4 f0 = *(const float4*)(p);
    float4 f1 = *(const float4*)(p + 4);
    float4 f2 = *(const float4*)(p + 16);
    float4 f3 = *(const float4*)(p + 20);
    union { v16bf v; __bf16 e[16]; } u;
    u.e[0]  = (__bf16)f0.x; u.e[1]  = (__bf16)f0.y; u.e[2]  = (__bf16)f0.z; u.e[3]  = (__bf16)f0.w;
    u.e[4]  = (__bf16)f1.x; u.e[5]  = (__bf16)f1.y; u.e[6]  = (__bf16)f1.z; u.e[7]  = (__bf16)f1.w;
    u.e[8]  = (__bf16)f2.x; u.e[9]  = (__bf16)f2.y; u.e[10] = (__bf16)f2.z; u.e[11] = (__bf16)f2.w;
    u.e[12] = (__bf16)f3.x; u.e[13] = (__bf16)f3.y; u.e[14] = (__bf16)f3.z; u.e[15] = (__bf16)f3.w;
    return u.v;
}

// A-fragment directly from a bf16 row-major MxK buffer.
__device__ __forceinline__ v16bf load_a_bf16frag(const __bf16* __restrict__ A, int ldk,
                                                 int row, int kt, int lane) {
    const __bf16* p = A + (size_t)row * ldk + kt + ((lane >> 4) << 3);
    v8bf lo = *(const v8bf*)(p);
    v8bf hi = *(const v8bf*)(p + 16);
    return __builtin_shufflevector(lo, hi, 0,1,2,3,4,5,6,7,8,9,10,11,12,13,14,15);
}

// B-fragment (32x16 KxN) from a W[n][k] row-major buffer (global).
__device__ __forceinline__ v16bf load_b_frag(const __bf16* __restrict__ W, int ldk,
                                             int ncol, int kt, int lane) {
    const __bf16* p = W + (size_t)(ncol + (lane & 15)) * ldk + kt + ((lane >> 4) << 3);
    v8bf lo = *(const v8bf*)(p);
    v8bf hi = *(const v8bf*)(p + 16);
    return __builtin_shufflevector(lo, hi, 0,1,2,3,4,5,6,7,8,9,10,11,12,13,14,15);
}

// B-fragment from an LDS-staged tile (row pitch KT bf16).
__device__ __forceinline__ v16bf lds_b_frag(const __bf16* sB, int trow, int kk, int lane) {
    const __bf16* p = sB + (trow + (lane & 15)) * KT + kk + ((lane >> 4) << 3);
    v8bf lo = *(const v8bf*)(p);
    v8bf hi = *(const v8bf*)(p + 16);
    return __builtin_shufflevector(lo, hi, 0,1,2,3,4,5,6,7,8,9,10,11,12,13,14,15);
}

// D tile (16x16 f32): lane col = lane&15, rows mbase + (lane>>4)*8 + j.
__device__ __forceinline__ void store_tile(float* __restrict__ C, int ldc,
                                           int mtile, int ntile, int lane, v8f acc) {
    int col   = ntile * 16 + (lane & 15);
    int rbase = mtile * 16 + ((lane >> 4) << 3);
#pragma unroll
    for (int j = 0; j < 8; ++j)
        C[(size_t)(rbase + j) * ldc + col] = acc[j];
}

// ---------------- K0: f32 -> bf16 weight conversion ----------------
__global__ void k_cvt_bf16(const float* __restrict__ src, __bf16* __restrict__ dst, int n) {
    int i = blockIdx.x * blockDim.x + threadIdx.x;
    if (i < n) dst[i] = (__bf16)src[i];
}

// ---------------- shared GEMM body with TDM-staged weights ----------------
// Block = 128 threads (4 waves); block covers 64 N-columns, each wave owns one
// 16-row mtile across all 4 ntiles. Weights for the block's 64 rows are DMA'd
// into LDS in KT-wide k-chunks, double-buffered: wave 0 issues
// TENSOR_LOAD_TO_LDS for chunk c+1, everyone computes chunk c from the other
// buffer, then s_wait_tensorcnt + barrier hands the new buffer over.
template <int KDIM, bool A_IS_BF16, bool ADD_RES>
__device__ __forceinline__ void gemm_tdm_body(const void* __restrict__ Aptr,
                                              const __bf16* __restrict__ Wb,
                                              const float* __restrict__ Res,
                                              float* __restrict__ Out, int ldo,
                                              __bf16* sB /*[2][64*KT]*/) {
    int lane  = threadIdx.x & 31;
    int wave  = threadIdx.x >> 5;
    int mtile = blockIdx.y * 4 + wave;
    int n0    = blockIdx.x * 64;
    int row   = mtile * 16 + (lane & 15);
    const __bf16* wbase = Wb + (size_t)n0 * KDIM;
    v8f acc[4];
#pragma unroll
    for (int t = 0; t < 4; ++t) acc[t] = (v8f){};

    const int NC = KDIM / KT;
#if USE_TDM
    if (wave == 0) {
        tdm_load_tile((unsigned)(size_t)&sB[0], wbase, KT, 64, KT, 64, KDIM);
        __builtin_amdgcn_s_wait_tensorcnt(0);
    }
    __syncthreads();
#else
    copy_tile_coop(&sB[0], wbase, 64, KT, KDIM, threadIdx.x, 128);
    __syncthreads();
#endif
    for (int c = 0; c < NC; ++c) {
        int buf = c & 1;
#if USE_TDM
        if (wave == 0 && c + 1 < NC)
            tdm_load_tile((unsigned)(size_t)&sB[(buf ^ 1) * 64 * KT],
                          wbase + (size_t)(c + 1) * KT, KT, 64, KT, 64, KDIM);
#else
        if (c + 1 < NC)
            copy_tile_coop(&sB[(buf ^ 1) * 64 * KT], wbase + (size_t)(c + 1) * KT,
                           64, KT, KDIM, threadIdx.x, 128);
#endif
#pragma unroll
        for (int ks = 0; ks < KT; ks += 32) {
            v16bf a;
            if (A_IS_BF16)
                a = load_a_bf16frag((const __bf16*)Aptr, KDIM, row, c * KT + ks, lane);
            else
                a = load_a_f32frag((const float*)Aptr, KDIM, row, c * KT + ks, lane);
#pragma unroll
            for (int t = 0; t < 4; ++t) {
                v16bf b = lds_b_frag(&sB[buf * 64 * KT], t * 16, ks, lane);
                acc[t] = wmma_bf16(a, b, acc[t]);
            }
        }
        __syncthreads();
#if USE_TDM
        if (wave == 0 && c + 1 < NC) __builtin_amdgcn_s_wait_tensorcnt(0);
        __syncthreads();
#endif
    }
#pragma unroll
    for (int t = 0; t < 4; ++t) {
        if (ADD_RES) {
            int col   = n0 + t * 16 + (lane & 15);
            int rbase = mtile * 16 + ((lane >> 4) << 3);
#pragma unroll
            for (int j = 0; j < 8; ++j) {
                size_t o = (size_t)(rbase + j) * ldo + col;
                Out[o] = acc[t][j] + Res[o];
            }
        } else {
            store_tile(Out, ldo, mtile, blockIdx.x * 4 + t, lane, acc[t]);
        }
    }
}

// ---------------- K1: xz = x @ W_in^T  (M=4096, K=1024, N=4096) ----------------
__global__ void k_inproj(const float* __restrict__ X, const __bf16* __restrict__ Wb,
                         float* __restrict__ XZ) {
    __shared__ __align__(128) __bf16 sB[2 * 64 * KT];
    gemm_tdm_body<DM, false, false>(X, Wb, nullptr, XZ, 2 * DI, sB);
}

// ---------------- K2: depthwise conv3 + SiLU ----------------
__global__ void k_conv_silu(const float* __restrict__ XZ, const float* __restrict__ cw,
                            const float* __restrict__ cb, float* __restrict__ Xact) {
    int idx = blockIdx.x * blockDim.x + threadIdx.x;   // over B*S*DI
    if (idx >= MROWS * DI) return;
    int e  = idx & (DI - 1);
    int bl = idx >> 11;                // DI = 2048
    int l  = bl & (SS - 1);
    float acc = cb[e];
    const float* base = XZ + (size_t)bl * (2 * DI) + e;
    float w0 = cw[e * 3 + 0], w1 = cw[e * 3 + 1], w2 = cw[e * 3 + 2];
    if (l > 0)      acc += w0 * base[-(2 * DI)];
    acc += w1 * base[0];
    if (l < SS - 1) acc += w2 * base[ (2 * DI)];
    Xact[idx] = acc / (1.f + __expf(-acc));            // silu
}

// ---------------- K3: dbc = x_act @ W_x^T  (M=4096, K=2048, N=96) ----------------
__global__ void k_dbc(const float* __restrict__ Xact, const __bf16* __restrict__ Wxb,
                      float* __restrict__ DBC, __bf16* __restrict__ DTRAWB) {
    const int K = DI, NLD = DR + 2 * NS;   // 96
    int lane  = threadIdx.x & 31;
    int wave  = threadIdx.x >> 5;
    int mtile = blockIdx.x * 4 + wave;
    int row   = mtile * 16 + (lane & 15);
    v8f acc[6];
#pragma unroll
    for (int t = 0; t < 6; ++t) acc[t] = (v8f){};
    for (int kt = 0; kt < K; kt += 32) {
        v16bf a = load_a_f32frag(Xact, K, row, kt, lane);
#pragma unroll
        for (int t = 0; t < 6; ++t) {
            v16bf b = load_b_frag(Wxb, K, t * 16, kt, lane);
            acc[t] = wmma_bf16(a, b, acc[t]);
        }
    }
#pragma unroll
    for (int t = 0; t < 6; ++t) {
        int col   = t * 16 + (lane & 15);
        int rbase = mtile * 16 + ((lane >> 4) << 3);
#pragma unroll
        for (int j = 0; j < 8; ++j) {
            float v = acc[t][j];
            DBC[(size_t)(rbase + j) * NLD + col] = v;
            if (col < DR) DTRAWB[(size_t)(rbase + j) * DR + col] = (__bf16)v;
        }
    }
}

// ------- K4: dt = softplus(dt_raw @ W_dt^T + b_dt)  (M=4096, K=64, N=2048) -------
__global__ void k_dt(const __bf16* __restrict__ DTRAWB, const __bf16* __restrict__ Wdtb,
                     const float* __restrict__ bdt, float* __restrict__ DT) {
    const int K = DR;
    int lane  = threadIdx.x & 31;
    int wave  = threadIdx.x >> 5;
    int mtile = blockIdx.y * 4 + wave;
    int nt0   = blockIdx.x * 4;
    int row   = mtile * 16 + (lane & 15);
    v8f acc[4];
#pragma unroll
    for (int t = 0; t < 4; ++t) acc[t] = (v8f){};
    for (int kt = 0; kt < K; kt += 32) {
        v16bf a = load_a_bf16frag(DTRAWB, K, row, kt, lane);
#pragma unroll
        for (int t = 0; t < 4; ++t) {
            v16bf b = load_b_frag(Wdtb, K, (nt0 + t) * 16, kt, lane);
            acc[t] = wmma_bf16(a, b, acc[t]);
        }
    }
#pragma unroll
    for (int t = 0; t < 4; ++t) {
        int col   = (nt0 + t) * 16 + (lane & 15);
        int rbase = mtile * 16 + ((lane >> 4) << 3);
        float bias = bdt[col];
#pragma unroll
        for (int j = 0; j < 8; ++j) {
            float v  = acc[t][j] + bias;
            float sp = (v > 20.f) ? v : log1pf(__expf(v));   // softplus
            DT[(size_t)(rbase + j) * DI + col] = sp;
        }
    }
}

// ---------------- K5: selective scan (sequential in L) ----------------
// 16 lanes per channel: lane n holds state n. grid = B*DI/16 blocks of 256.
__global__ void k_scan(const float* __restrict__ DT, const float* __restrict__ Xact,
                       const float* __restrict__ DBC, const float* __restrict__ XZ,
                       const float* __restrict__ Alog, const float* __restrict__ Dp,
                       __bf16* __restrict__ YB) {
    const int NLD = DR + 2 * NS;
    int b = blockIdx.x >> 7;                               // 128 blocks per batch
    int e = ((blockIdx.x & 127) << 4) + (threadIdx.x >> 4);
    int n = threadIdx.x & 15;
    float An  = -__expf(Alog[e * NS + n]);
    float Dpe = Dp[e];
    float h = 0.f;
    size_t r = (size_t)b * SS;
    for (int l = 0; l < SS; ++l, ++r) {
        float dtv = DT[r * DI + e];
        float xv  = Xact[r * DI + e];
        float Bn  = DBC[r * NLD + DR + n];
        float Cn  = DBC[r * NLD + DR + NS + n];
        if (l + 1 < SS) {
            __builtin_prefetch(&DT[(r + 1) * DI + e], 0, 3);
            __builtin_prefetch(&Xact[(r + 1) * DI + e], 0, 3);
        }
        h = __expf(dtv * An) * h + dtv * Bn * xv;
        float c = h * Cn;
        c += __shfl_xor(c, 1);
        c += __shfl_xor(c, 2);
        c += __shfl_xor(c, 4);
        c += __shfl_xor(c, 8);
        if (n == 0) {
            float z = XZ[r * (2 * DI) + DI + e];
            float g = z / (1.f + __expf(-z));              // silu(z)
            float comb = (c + xv * Dpe) * g;
            YB[r * DI + e] = (__bf16)comb;
        }
    }
}

// ------- K6: out = y @ W_out^T + residual  (M=4096, K=2048, N=1024) -------
__global__ void k_outproj(const __bf16* __restrict__ YB, const __bf16* __restrict__ Wob,
                          const float* __restrict__ Xres, float* __restrict__ OUT) {
    __shared__ __align__(128) __bf16 sB[2 * 64 * KT];
    gemm_tdm_body<DI, true, true>(YB, Wob, Xres, OUT, DM, sB);
}

// ---------------- K7: in-place LayerNorm over DM ----------------
__global__ void k_layernorm(float* __restrict__ OUT, const float* __restrict__ gamma,
                            const float* __restrict__ beta) {
    __shared__ float s1[256], s2[256];
    int row = blockIdx.x;
    float* p = OUT + (size_t)row * DM;
    float v[4]; float s = 0.f, sq = 0.f;
#pragma unroll
    for (int i = 0; i < 4; ++i) {
        v[i] = p[threadIdx.x + i * 256];
        s += v[i]; sq += v[i] * v[i];
    }
    s1[threadIdx.x] = s; s2[threadIdx.x] = sq;
    __syncthreads();
    for (int off = 128; off > 0; off >>= 1) {
        if ((int)threadIdx.x < off) {
            s1[threadIdx.x] += s1[threadIdx.x + off];
            s2[threadIdx.x] += s2[threadIdx.x + off];
        }
        __syncthreads();
    }
    float mu   = s1[0] * (1.f / DM);
    float var  = s2[0] * (1.f / DM) - mu * mu;
    float rstd = rsqrtf(var + EPSL);
#pragma unroll
    for (int i = 0; i < 4; ++i) {
        int c = threadIdx.x + i * 256;
        p[c] = (v[i] - mu) * rstd * gamma[c] + beta[c];
    }
}

// ---------------- host launcher ----------------
extern "C" void kernel_launch(void* const* d_in, const int* in_sizes, int n_in,
                              void* d_out, int out_size, void* d_ws, size_t ws_size,
                              hipStream_t stream) {
    (void)in_sizes; (void)n_in; (void)out_size; (void)ws_size;
    const float* x      = (const float*)d_in[0];
    const float* W_in   = (const float*)d_in[1];
    const float* conv_w = (const float*)d_in[2];
    const float* conv_b = (const float*)d_in[3];
    const float* W_x    = (const float*)d_in[4];
    const float* W_dt   = (const float*)d_in[5];
    const float* b_dt   = (const float*)d_in[6];
    const float* A_log  = (const float*)d_in[7];
    const float* D_par  = (const float*)d_in[8];
    const float* W_out  = (const float*)d_in[9];
    const float* gamma  = (const float*)d_in[10];
    const float* beta   = (const float*)d_in[11];
    float* out = (float*)d_out;

    // workspace layout (all offsets 256B aligned)
    unsigned char* ws = (unsigned char*)d_ws;
    size_t off = 0;
    __bf16* Wb_in  = (__bf16*)(ws + off); off += (size_t)(2 * DI) * DM * 2;      // 8 MB
    __bf16* Wb_x   = (__bf16*)(ws + off); off += (size_t)(DR + 2 * NS) * DI * 2; // 384 KB
    __bf16* Wb_dt  = (__bf16*)(ws + off); off += (size_t)DI * DR * 2;            // 256 KB
    __bf16* Wb_out = (__bf16*)(ws + off); off += (size_t)DM * DI * 2;            // 4 MB
    float*  XZ     = (float*)(ws + off);  off += (size_t)MROWS * (2 * DI) * 4;   // 64 MB
    float*  Xact   = (float*)(ws + off);  off += (size_t)MROWS * DI * 4;         // 32 MB
    float*  DBC    = (float*)(ws + off);  off += (size_t)MROWS * (DR + 2 * NS) * 4;
    __bf16* DTRAWB = (__bf16*)(ws + off); off += (size_t)MROWS * DR * 2;
    float*  DT     = (float*)(ws + off);  off += (size_t)MROWS * DI * 4;         // 32 MB
    __bf16* YB     = (__bf16*)(ws + off); off += (size_t)MROWS * DI * 2;         // 16 MB

    // K0: weight conversions
    {
        int n;
        n = 2 * DI * DM;        k_cvt_bf16<<<(n + 255) / 256, 256, 0, stream>>>(W_in,  Wb_in,  n);
        n = (DR + 2 * NS) * DI; k_cvt_bf16<<<(n + 255) / 256, 256, 0, stream>>>(W_x,   Wb_x,   n);
        n = DI * DR;            k_cvt_bf16<<<(n + 255) / 256, 256, 0, stream>>>(W_dt,  Wb_dt,  n);
        n = DM * DI;            k_cvt_bf16<<<(n + 255) / 256, 256, 0, stream>>>(W_out, Wb_out, n);
    }
    // K1: in-projection GEMM (TDM-staged weights)
    k_inproj<<<dim3((2 * DI) / 64, MROWS / 64), 128, 0, stream>>>(x, Wb_in, XZ);
    // K2: conv3 + silu
    k_conv_silu<<<(MROWS * DI) / 256, 256, 0, stream>>>(XZ, conv_w, conv_b, Xact);
    // K3: dbc GEMM (+ bf16 copy of dt_raw)
    k_dbc<<<MROWS / 64, 128, 0, stream>>>(Xact, Wb_x, DBC, DTRAWB);
    // K4: dt GEMM + softplus
    k_dt<<<dim3(DI / 64, MROWS / 64), 128, 0, stream>>>(DTRAWB, Wb_dt, b_dt, DT);
    // K5: selective scan (fused D-skip + silu gate)
    k_scan<<<(BB * DI) / 16, 256, 0, stream>>>(DT, Xact, DBC, XZ, A_log, D_par, YB);
    // K6: out-projection GEMM + residual (TDM-staged weights)
    k_outproj<<<dim3(DM / 64, MROWS / 64), 128, 0, stream>>>(YB, Wb_out, x, out);
    // K7: in-place LayerNorm
    k_layernorm<<<MROWS, 256, 0, stream>>>(out, gamma, beta);
}
